// MultiHeadAttention_38182259261994
// MI455X (gfx1250) — matrix operand order repaired and checked by reading, not compiled, not run on hardware
//
#include <hip/hip_runtime.h>
#include <hip/hip_bf16.h>

// ---------------------------------------------------------------------------
// Causal multi-head attention forward for MI455X (gfx1250), bf16 WMMA path.
//   stage 1: Q/K/V = x @ W{q,k,v}           (WMMA bf16 GEMM, bf16 out, [B,H,T,64])
//   stage 2: flash attention per (b,h)      (WMMA scores + WMMA P@V, f32 online softmax)
//   stage 3: out = ctx @ Wo + bo            (WMMA bf16 GEMM, f32 out)
//
// WMMA operand layouts as used here (A/B k order is this code's own, not the ISA table order):
//   A (16x32 bf16): lane    -> row=lane&15, K = half*16 .. half*16+15 (contiguous)
//   B (32x16 bf16): lane    -> col=lane&15, K = half*16 .. half*16+15 (contiguous)
//   C/D (16x16 f32): VGPR r -> row = half*8 + r, col = lane&15
// ---------------------------------------------------------------------------

typedef __bf16 bf16_t;
typedef __attribute__((ext_vector_type(16))) __bf16 v16bf;
typedef __attribute__((ext_vector_type(8)))  __bf16 v8bf;
typedef __attribute__((ext_vector_type(4)))  __bf16 v4bf;
typedef __attribute__((ext_vector_type(8)))  float  v8f;
typedef __attribute__((ext_vector_type(4)))  float  v4f;

#define D_MODEL  1024
#define N_HEADS  16
#define HEAD_DIM 64
#define SEQ      2048
#define BATCH    2
#define M_TOTAL  (BATCH * SEQ)   // 4096 token rows

static __device__ __forceinline__ v16bf combine8(v8bf lo, v8bf hi) {
    return __builtin_shufflevector(lo, hi, 0,1,2,3,4,5,6,7,8,9,10,11,12,13,14,15);
}

static __device__ __forceinline__ v8f wmma_bf16(v16bf a, v16bf b, v8f c) {
    // v_wmma_f32_16x16x32_bf16: (neg_a, A, neg_b, B, c_mod, C, reuse_a, reuse_b)
    return __builtin_amdgcn_wmma_f32_16x16x32_bf16(false, a, false, b, (short)0, c,
                                                   false, false);
}

// A-operand fragment: row-major LDS tile, row stride `STRIDE` bf16 elements.
template <int STRIDE>
static __device__ __forceinline__ v16bf load_a_frag(const bf16_t* base, int row,
                                                    int koff, int kb) {
    const bf16_t* p = base + (size_t)row * STRIDE + koff;
    v8bf lo = *(const v8bf*)(p + 2 * kb);
    v8bf hi = *(const v8bf*)(p + 2 * kb + 8);
    return combine8(lo, hi);
}

// B-operand fragment from a K-contiguous (transposed) LDS tile: tile[col][k],
// per-lane 16 contiguous K values starting at half*16.
template <int STRIDE>
static __device__ __forceinline__ v16bf load_b_frag(const bf16_t* base, int col,
                                                    int koff, int half) {
    const bf16_t* p = base + (size_t)col * STRIDE + koff + half * 16;
    v8bf lo = *(const v8bf*)(p);
    v8bf hi = *(const v8bf*)(p + 8);
    return combine8(lo, hi);
}

// ---------------------------------------------------------------------------
// Stage 1 / Stage 3 GEMM: 128x128 block tile, K-step 64, 256 threads (8 waves).
// Wave grid 2x4 -> each wave computes a 64x32 region = 4x2 WMMA tiles.
// MODE 0: A = x (f32), out = bf16 in head-split [B,H,T,64] layout (QKV).
// MODE 1: A = ctx (bf16), out = f32 row-major + bias (output projection).
// ---------------------------------------------------------------------------
template <int MODE>
__global__ __launch_bounds__(256)
void mha_gemm128(const void* __restrict__ Ap, const float* __restrict__ Wp,
                 const float* __restrict__ bias, void* __restrict__ outp) {
    constexpr int K = D_MODEL;
    constexpr int N = D_MODEL;

    __shared__ bf16_t As[128][72];   // A tile 128x64, row stride 144B (16B aligned)
    __shared__ bf16_t Bs[128][72];   // W tile 64x128 TRANSPOSED: Bs[n][k]

    const int tid  = threadIdx.x;
    const int lane = tid & 31;
    const int wave = tid >> 5;       // 0..7
    const int wm   = wave >> 2;      // 0..1
    const int wn   = wave & 3;       // 0..3
    const int half = lane >> 4;      // 0/1
    const int l16  = lane & 15;
    const int kb   = half * 8;       // A-operand K sub-group base

    const int n0 = blockIdx.x * 128;
    const int m0 = blockIdx.y * 128;

    v8f acc[4][2];
#pragma unroll
    for (int tm = 0; tm < 4; ++tm)
#pragma unroll
        for (int tn = 0; tn < 2; ++tn) acc[tm][tn] = {};

    for (int k0 = 0; k0 < K; k0 += 64) {
        __syncthreads();
        // ---- stage A tile (128 x 64) into LDS as bf16 ----
        if (MODE == 0) {
            const float* A = (const float*)Ap;
#pragma unroll
            for (int j = 0; j < 8; ++j) {
                int idx = j * 256 + tid;          // 2048 float4 chunks
                int row = idx >> 4;
                int c4  = (idx & 15) * 4;
                v4f v = *(const v4f*)(A + (size_t)(m0 + row) * K + k0 + c4);
                v4bf bv;
#pragma unroll
                for (int e = 0; e < 4; ++e) bv[e] = (bf16_t)v[e];
                *(v4bf*)&As[row][c4] = bv;        // 8B LDS store
            }
        } else {
            const bf16_t* A = (const bf16_t*)Ap;
#pragma unroll
            for (int j = 0; j < 4; ++j) {
                int idx = j * 256 + tid;          // 1024 v8bf chunks
                int row = idx >> 3;
                int seg = (idx & 7) * 8;
                *(v8bf*)&As[row][seg] =
                    *(const v8bf*)(A + (size_t)(m0 + row) * K + k0 + seg);
            }
        }
        // ---- stage W tile (64 x 128) transposed into LDS as bf16 ----
#pragma unroll
        for (int j = 0; j < 8; ++j) {
            int idx = j * 256 + tid;              // 2048 float4 chunks
            int kr  = idx >> 5;
            int n4  = (idx & 31) * 4;
            v4f v = *(const v4f*)(Wp + (size_t)(k0 + kr) * N + n0 + n4);
#pragma unroll
            for (int e = 0; e < 4; ++e) Bs[n4 + e][kr] = (bf16_t)v[e];
        }
        __syncthreads();

        // prefetch next weight tile into cache while we do math
        if (k0 + 64 < K)
            __builtin_prefetch(Wp + (size_t)(k0 + 64 + (tid >> 2)) * N + n0 + (tid & 3) * 32,
                               0, 3);

        // ---- WMMA: 2 K-chunks x (4 A-frags, 2 B-frags, 8 MMAs) ----
#pragma unroll
        for (int c = 0; c < 2; ++c) {
            v16bf afr[4];
#pragma unroll
            for (int tm = 0; tm < 4; ++tm)
                afr[tm] = load_a_frag<72>(&As[0][0], wm * 64 + tm * 16 + l16,
                                          c * 32, kb);
#pragma unroll
            for (int tn = 0; tn < 2; ++tn) {
                v16bf bfr = load_b_frag<72>(&Bs[0][0], wn * 32 + tn * 16 + l16,
                                            c * 32, half);
#pragma unroll
                for (int tm = 0; tm < 4; ++tm)
                    acc[tm][tn] = wmma_bf16(afr[tm], bfr, acc[tm][tn]);
            }
        }
    }

    // ---- epilogue ----
#pragma unroll
    for (int tm = 0; tm < 4; ++tm)
#pragma unroll
        for (int tn = 0; tn < 2; ++tn) {
            int gcol = n0 + wn * 32 + tn * 16 + l16;
#pragma unroll
            for (int r = 0; r < 8; ++r) {
                int grow = m0 + wm * 64 + tm * 16 + half * 8 + r;
                float v = acc[tm][tn][r];
                if (MODE == 0) {
                    bf16_t* out = (bf16_t*)outp;
                    int bb = grow >> 11, t = grow & (SEQ - 1);
                    int h = gcol >> 6, dh = gcol & (HEAD_DIM - 1);
                    out[(((size_t)(bb * N_HEADS + h) * SEQ + t) << 6) + dh] = (bf16_t)v;
                } else {
                    float* out = (float*)outp;
                    out[(size_t)grow * N + gcol] = v + bias[gcol];
                }
            }
        }
}

// ---------------------------------------------------------------------------
// Stage 2: causal flash attention. Block = one (b,h) x 128 queries, 8 waves.
// Each wave owns 16 query rows; loops over 64-key LDS-staged blocks.
// ---------------------------------------------------------------------------
__global__ __launch_bounds__(256)
void mha_flash_attn(const bf16_t* __restrict__ Q, const bf16_t* __restrict__ Kq,
                    const bf16_t* __restrict__ V, bf16_t* __restrict__ ctx) {
    __shared__ bf16_t Qs[128][72];       // 128 queries x 64 dh (+pad)
    __shared__ bf16_t Ks[64][72];        // 64 keys x 64 dh (B-operand: K^T, col=key)
    __shared__ bf16_t Vt[64][72];        // V transposed: [dh][key] (B-operand, col=dh)
    __shared__ bf16_t Ps[8][16][40];     // per-wave P relayout scratch (A-operand)

    const int tid  = threadIdx.x;
    const int lane = tid & 31;
    const int wave = tid >> 5;
    const int half = lane >> 4;
    const int l16  = lane & 15;
    const int kb   = half * 8;

    const int bh    = blockIdx.y;        // 0..31
    const int bidx  = bh >> 4;
    const int h     = bh & 15;
    const int qbase = blockIdx.x * 128;

    const bf16_t* Qh = Q  + (size_t)bh * SEQ * HEAD_DIM;
    const bf16_t* Kh = Kq + (size_t)bh * SEQ * HEAD_DIM;
    const bf16_t* Vh = V  + (size_t)bh * SEQ * HEAD_DIM;

    // ---- stage Q block ----
#pragma unroll
    for (int j = 0; j < 4; ++j) {
        int idx = j * 256 + tid;
        int row = idx >> 3, seg = (idx & 7) * 8;
        *(v8bf*)&Qs[row][seg] = *(const v8bf*)(Qh + (size_t)(qbase + row) * HEAD_DIM + seg);
    }
    __syncthreads();

    v16bf qfr[2];
#pragma unroll
    for (int c = 0; c < 2; ++c)
        qfr[c] = load_a_frag<72>(&Qs[0][0], wave * 16 + l16, c * 32, kb);

    v8f o[4];
#pragma unroll
    for (int d = 0; d < 4; ++d) o[d] = {};
    float m_i[8], l_i[8];
#pragma unroll
    for (int r = 0; r < 8; ++r) { m_i[r] = -1e30f; l_i[r] = 0.0f; }

    const int qrow_base = qbase + wave * 16 + half * 8;
    const int nkb = (qbase >> 6) + 2;    // causal: key blocks up to qbase+127

    for (int kblk = 0; kblk < nkb; ++kblk) {
        const int krow0 = kblk * 64;
        __syncthreads();
        // ---- stage K (row-major) and V (transposed) tiles ----
#pragma unroll
        for (int j = 0; j < 2; ++j) {
            int idx = j * 256 + tid;
            int row = idx >> 3, seg = (idx & 7) * 8;
            *(v8bf*)&Ks[row][seg] =
                *(const v8bf*)(Kh + (size_t)(krow0 + row) * HEAD_DIM + seg);
            v8bf vv = *(const v8bf*)(Vh + (size_t)(krow0 + row) * HEAD_DIM + seg);
#pragma unroll
            for (int e = 0; e < 8; ++e) Vt[seg + e][row] = vv[e];
        }
        __syncthreads();

        // ---- scores: S = Q @ K^T for 4 key sub-tiles of 16 ----
        v8f s[4];
#pragma unroll
        for (int nt = 0; nt < 4; ++nt) {
            v8f sacc = {};
#pragma unroll
            for (int c = 0; c < 2; ++c) {
                v16bf kfr = load_b_frag<72>(&Ks[0][0], nt * 16 + l16, c * 32, half);
                sacc = wmma_bf16(qfr[c], kfr, sacc);
            }
            s[nt] = sacc;
        }

        // ---- scale + causal mask ----
        const float scale = 0.125f;       // 1/sqrt(64)
#pragma unroll
        for (int nt = 0; nt < 4; ++nt) {
            int key = krow0 + nt * 16 + l16;
#pragma unroll
            for (int r = 0; r < 8; ++r) {
                float v = s[nt][r] * scale;
                s[nt][r] = (key <= qrow_base + r) ? v : -1e30f;
            }
        }

        // ---- online softmax (rows live in 16-lane halves) ----
        float mnew[8], alpha[8];
#pragma unroll
        for (int r = 0; r < 8; ++r) {
            float mx = fmaxf(fmaxf(s[0][r], s[1][r]), fmaxf(s[2][r], s[3][r]));
#pragma unroll
            for (int msk = 1; msk <= 8; msk <<= 1)
                mx = fmaxf(mx, __shfl_xor(mx, msk, 32));
            mnew[r]  = fmaxf(m_i[r], mx);
            alpha[r] = __expf(m_i[r] - mnew[r]);
            m_i[r]   = mnew[r];
        }
#pragma unroll
        for (int nt = 0; nt < 4; ++nt)
#pragma unroll
            for (int r = 0; r < 8; ++r)
                s[nt][r] = __expf(s[nt][r] - mnew[r]);
#pragma unroll
        for (int r = 0; r < 8; ++r) {
            float rs = s[0][r] + s[1][r] + s[2][r] + s[3][r];
#pragma unroll
            for (int msk = 1; msk <= 8; msk <<= 1)
                rs += __shfl_xor(rs, msk, 32);
            l_i[r] = l_i[r] * alpha[r] + rs;
        }
#pragma unroll
        for (int d = 0; d < 4; ++d)
#pragma unroll
            for (int r = 0; r < 8; ++r) o[d][r] *= alpha[r];

        // ---- relayout P (C-frag f32 -> A-frag bf16) through per-wave LDS ----
#pragma unroll
        for (int nt = 0; nt < 4; ++nt)
#pragma unroll
            for (int r = 0; r < 8; ++r)
                Ps[wave][half * 8 + r][nt * 16 + l16] = (bf16_t)s[nt][r];
        asm volatile("s_wait_dscnt 0x0" ::: "memory");  // wave-local LDS RAW fence

        // ---- O += P @ V ----
#pragma unroll
        for (int c = 0; c < 2; ++c) {
            v16bf pfr = load_a_frag<40>(&Ps[wave][0][0], l16, c * 32, kb);
#pragma unroll
            for (int d = 0; d < 4; ++d) {
                v16bf vfr = load_b_frag<72>(&Vt[0][0], d * 16 + l16, c * 32, half);
                o[d] = wmma_bf16(pfr, vfr, o[d]);
            }
        }
    }

    // ---- normalize and store ctx in [B,T,C] (heads re-interleaved) ----
#pragma unroll
    for (int r = 0; r < 8; ++r) {
        float inv = 1.0f / l_i[r];
        int q = qrow_base + r;
        size_t rowbase = ((size_t)(bidx * SEQ + q)) * D_MODEL + h * HEAD_DIM;
#pragma unroll
        for (int d = 0; d < 4; ++d)
            ctx[rowbase + d * 16 + l16] = (bf16_t)(o[d][r] * inv);
    }
}

// ---------------------------------------------------------------------------
extern "C" void kernel_launch(void* const* d_in, const int* in_sizes, int n_in,
                              void* d_out, int out_size, void* d_ws, size_t ws_size,
                              hipStream_t stream) {
    const float* x  = (const float*)d_in[0];
    const float* Wq = (const float*)d_in[1];
    const float* Wk = (const float*)d_in[2];
    const float* Wv = (const float*)d_in[3];
    const float* Wo = (const float*)d_in[4];
    const float* bo = (const float*)d_in[5];

    const size_t mat = (size_t)M_TOTAL * D_MODEL;   // 4M elements
    bf16_t* qb   = (bf16_t*)d_ws;
    bf16_t* kbuf = qb + mat;
    bf16_t* vbuf = kbuf + mat;
    bf16_t* ctx  = vbuf + mat;                      // total 32 MB bf16 scratch

    dim3 ggrid(D_MODEL / 128, M_TOTAL / 128);       // (8, 32)

    mha_gemm128<0><<<ggrid, 256, 0, stream>>>(x, Wq, nullptr, qb);
    mha_gemm128<0><<<ggrid, 256, 0, stream>>>(x, Wk, nullptr, kbuf);
    mha_gemm128<0><<<ggrid, 256, 0, stream>>>(x, Wv, nullptr, vbuf);

    mha_flash_attn<<<dim3(SEQ / 128, BATCH * N_HEADS), 256, 0, stream>>>(qb, kbuf, vbuf, ctx);

    mha_gemm128<1><<<ggrid, 256, 0, stream>>>(ctx, Wo, bo, d_out);
}
